// UNet_3590592659678
// MI455X (gfx1250) — matrix operand order repaired and checked by reading, not compiled, not run on hardware
//
#include <hip/hip_runtime.h>
#include <cstdint>
#include <cstddef>

// ---------------- types ----------------
typedef _Float16 half_t;
typedef __attribute__((ext_vector_type(8)))  _Float16 v8h;
typedef __attribute__((ext_vector_type(16))) _Float16 v16h;
typedef __attribute__((ext_vector_type(8)))  float    v8f;

#define GSTRIDE(i, tot) \
  for (size_t i = (size_t)blockIdx.x * blockDim.x + threadIdx.x; i < (tot); \
       i += (size_t)gridDim.x * blockDim.x)

// =====================================================================
// Weight repack: fp32 conv / conv-transpose weights -> f16 GEMM A [Mp][Kp]
// Rows [Cout, Mp) and K slots [Cin*ks*ks, Kp) are zero-filled so the GEMM
// never needs per-lane bounds predication on its A loads.
// transposed=1: Wsrc is [Cin][Cout][ks][ks], spatially flipped (ConvT k3 s1 p1)
// =====================================================================
__global__ void k_wrepack(const float* __restrict__ Wsrc, half_t* __restrict__ A,
                          int Cout, int Mp, int Cin, int ks, int Kp, int transposed) {
  size_t tot = (size_t)Mp * Kp;
  int ks2 = ks * ks;
  GSTRIDE(idx, tot) {
    int o = (int)(idx / Kp);
    int k = (int)(idx % Kp);
    float v = 0.f;
    if (o < Cout && k < Cin * ks2) {
      int c = k / ks2, r = k % ks2, ky = r / ks, kx = r % ks;
      if (transposed)
        v = Wsrc[(((size_t)c * Cout + o) * ks + (ks - 1 - ky)) * ks + (ks - 1 - kx)];
      else
        v = Wsrc[(((size_t)o * Cin + c) * ks + ky) * ks + kx];
    }
    A[idx] = (half_t)v;
  }
}

// =====================================================================
// im2col, K-contiguous layout: col[n][k], n = b*H*W + y*W + x, k in [0,Kp)
// (K contiguous per pixel -> the GEMM B fragment is one 32-byte load)
// =====================================================================
__global__ void k_im2col(const float* __restrict__ X, half_t* __restrict__ col,
                         int B, int C, int H, int W, int ks, int pad, int Kp) {
  size_t Ncol = (size_t)B * H * W;
  size_t tot = Ncol * Kp;
  int ks2 = ks * ks, HW = H * W;
  GSTRIDE(idx, tot) {
    size_t n = idx / Kp;
    int k = (int)(idx % Kp);
    float v = 0.f;
    if (k < C * ks2) {
      int c = k / ks2, r = k % ks2, ky = r / ks, kx = r % ks;
      int b = (int)(n / HW), hw = (int)(n % HW);
      int y = hw / W, x = hw % W;
      int yy = y + ky - pad, xx = x + kx - pad;
      if (yy >= 0 && yy < H && xx >= 0 && xx < W)
        v = X[(((size_t)b * C + c) * H + yy) * W + xx];
    }
    col[idx] = (half_t)v;
  }
}

// =====================================================================
// Implicit GEMM: Y(NCHW fp32) = act( A[Mp][Kp] * col[Ncol][Kp]^T + bias )
// One wave computes a 16x32 output slab: shared A fragment, two
// independent v_wmma_f32_16x16x32_f16 accumulator chains per K panel
// (independent back-to-back WMMAs avoid the D->A/B RAW NOP penalty).
// act: 0=none 1=relu 2=tanh.  Ncol/16 is even for every layer here.
// =====================================================================
__global__ __launch_bounds__(256)
void k_gemm_wmma(const half_t* __restrict__ A, const half_t* __restrict__ Bm,
                 float* __restrict__ Y, const float* __restrict__ bias,
                 int M, int Ncol, int HW, int Kp, int act) {
  const int ntn2 = Ncol >> 5;                 // pairs of 16-col tiles
  const int ntm  = (M + 15) >> 4;
  int unit = blockIdx.x * (blockDim.x >> 5) + (threadIdx.x >> 5);
  if (unit >= ntm * ntn2) return;             // wave-uniform: EXEC stays all-ones
  const int tm = (unit / ntn2) << 4;
  const int tn = (unit % ntn2) << 5;
  const int lane = threadIdx.x & 31;
  const int lr = lane & 15;                   // row-in-tile (A) / col-in-tile (B,C)
  const int hi = lane >> 4;

  const half_t* arow  = A  + (size_t)(tm + lr) * Kp;       // Mp-padded: no guard
  const half_t* brow0 = Bm + (size_t)(tn + lr) * Kp;
  const half_t* brow1 = Bm + (size_t)(tn + 16 + lr) * Kp;

  v8f acc0 = {}, acc1 = {};
  for (int k0 = 0; k0 < Kp; k0 += 32) {
    // A 16x32 f16 fragment: lane holds contiguous K run [k0+hi*16, +16)
    v8h alo = *(const v8h*)(arow + k0 + (hi << 4));
    v8h ahi = *(const v8h*)(arow + k0 + 8 + (hi << 4));
    v16h af = __builtin_shufflevector(alo, ahi,
                                      0, 1, 2, 3, 4, 5, 6, 7,
                                      8, 9, 10, 11, 12, 13, 14, 15);
    // B 32x16 fragments: lane holds contiguous K run [k0+hi*16, +16) of its column
    v16h bf0 = *(const v16h*)(brow0 + k0 + (hi << 4));
    v16h bf1 = *(const v16h*)(brow1 + k0 + (hi << 4));
    if (k0 + 32 < Kp) {                       // global_prefetch_b8: next K panel
      __builtin_prefetch(brow0 + k0 + 32, 0, 1);
      __builtin_prefetch(brow1 + k0 + 32, 0, 1);
    }
    acc0 = __builtin_amdgcn_wmma_f32_16x16x32_f16(false, af, false, bf0,
                                                  (short)0, acc0, false, false);
    acc1 = __builtin_amdgcn_wmma_f32_16x16x32_f16(false, af, false, bf1,
                                                  (short)0, acc1, false, false);
  }
  // C/D layout: VGPR r -> row = r + (lane<16 ? 0 : 8), col = lane&15
#pragma unroll
  for (int r = 0; r < 8; ++r) {
    int row = tm + r + (hi << 3);
    if (row < M) {
      float bv = bias[row];
      int col0 = tn + lr, col1 = tn + 16 + lr;
      int b0 = col0 / HW, hw0 = col0 - b0 * HW;
      int b1 = col1 / HW, hw1 = col1 - b1 * HW;
      float v0 = acc0[r] + bv, v1 = acc1[r] + bv;
      if (act == 1)      { v0 = fmaxf(v0, 0.f); v1 = fmaxf(v1, 0.f); }
      else if (act == 2) { v0 = tanhf(v0);      v1 = tanhf(v1); }
      Y[((size_t)b0 * M + row) * HW + hw0] = v0;
      Y[((size_t)b1 * M + row) * HW + hw1] = v1;
    }
  }
}

// =====================================================================
// BatchNorm: stats (one block per channel) + apply (optionally fused relu)
// =====================================================================
__global__ __launch_bounds__(256)
void k_bn_stats(const float* __restrict__ X, float* __restrict__ mean,
                float* __restrict__ var, int B, int C, int HW) {
  int c = blockIdx.x;
  int n = B * HW;
  float s = 0.f, s2 = 0.f;
  for (int i = threadIdx.x; i < n; i += blockDim.x) {
    int b = i / HW, hw = i - b * HW;
    float v = X[((size_t)b * C + c) * HW + hw];
    s += v; s2 += v * v;
  }
  __shared__ float sh[256], sh2[256];
  sh[threadIdx.x] = s; sh2[threadIdx.x] = s2;
  __syncthreads();
  for (int st = 128; st > 0; st >>= 1) {
    if (threadIdx.x < st) {
      sh[threadIdx.x] += sh[threadIdx.x + st];
      sh2[threadIdx.x] += sh2[threadIdx.x + st];
    }
    __syncthreads();
  }
  if (threadIdx.x == 0) {
    float m = sh[0] / n;
    mean[c] = m;
    var[c] = sh2[0] / n - m * m;   // biased var, matches jnp.var
  }
}

__global__ void k_bn_apply(const float* __restrict__ X, float* __restrict__ Y,
                           const float* __restrict__ mean, const float* __restrict__ var,
                           const float* __restrict__ g, const float* __restrict__ bta,
                           int B, int C, int HW, int relu) {
  size_t tot = (size_t)B * C * HW;
  GSTRIDE(i, tot) {
    int c = (int)((i / HW) % C);
    float v = (X[i] - mean[c]) * rsqrtf(var[c] + 1e-5f) * g[c] + bta[c];
    if (relu) v = fmaxf(v, 0.f);
    Y[i] = v;
  }
}

// =====================================================================
// maxpool 2x2, bilinear up2 (align_corners), channel concat, sigmoid gate
// =====================================================================
__global__ void k_maxpool2(const float* __restrict__ X, float* __restrict__ Y,
                           int B, int C, int H, int W) {
  int Ho = H >> 1, Wo = W >> 1;
  size_t tot = (size_t)B * C * Ho * Wo;
  GSTRIDE(i, tot) {
    int wo = (int)(i % Wo);
    int ho = (int)((i / Wo) % Ho);
    size_t bc = i / ((size_t)Wo * Ho);
    const float* p = X + (bc * H + 2 * ho) * W + 2 * wo;
    Y[i] = fmaxf(fmaxf(p[0], p[1]), fmaxf(p[W], p[W + 1]));
  }
}

__global__ void k_up2(const float* __restrict__ X, float* __restrict__ Y,
                      int B, int C, int H, int W) {
  int Ho = 2 * H, Wo = 2 * W;
  size_t tot = (size_t)B * C * Ho * Wo;
  GSTRIDE(i, tot) {
    int xo = (int)(i % Wo);
    int yo = (int)((i / Wo) % Ho);
    size_t bc = i / ((size_t)Wo * Ho);
    float py = yo * (float)(H - 1) / (float)(Ho - 1);
    float px = xo * (float)(W - 1) / (float)(Wo - 1);
    int y0 = (int)py, x0 = (int)px;
    int y1 = (y0 + 1 < H) ? y0 + 1 : H - 1;
    int x1 = (x0 + 1 < W) ? x0 + 1 : W - 1;
    float wy = py - y0, wx = px - x0;
    const float* p = X + bc * H * W;
    float v = (1.f - wy) * ((1.f - wx) * p[(size_t)y0 * W + x0] + wx * p[(size_t)y0 * W + x1])
            +        wy  * ((1.f - wx) * p[(size_t)y1 * W + x0] + wx * p[(size_t)y1 * W + x1]);
    Y[i] = v;
  }
}

__global__ void k_concat(const float* __restrict__ A, const float* __restrict__ Bt,
                         float* __restrict__ Y, int B, int C1, int C2, int HW) {
  int C = C1 + C2;
  size_t tot = (size_t)B * C * HW;
  GSTRIDE(i, tot) {
    int hw = (int)(i % HW);
    int c = (int)((i / HW) % C);
    int b = (int)(i / ((size_t)HW * C));
    Y[i] = (c < C1) ? A[((size_t)b * C1 + c) * HW + hw]
                    : Bt[((size_t)b * C2 + (c - C1)) * HW + hw];
  }
}

// res_sdn merge: S=[B,2*oc,HW] (sdn_out | gate), Cn=[B,oc,HW]
__global__ void k_gate(const float* __restrict__ S, const float* __restrict__ Cn,
                       float* __restrict__ Y, int B, int oc, int HW) {
  size_t tot = (size_t)B * oc * HW;
  GSTRIDE(i, tot) {
    int hw = (int)(i % HW);
    int c = (int)((i / HW) % oc);
    int b = (int)(i / ((size_t)HW * oc));
    float sdn = S[((size_t)b * 2 * oc + c) * HW + hw];
    float gt  = S[((size_t)b * 2 * oc + oc + c) * HW + hw];
    float g = 1.f / (1.f + expf(-gt));
    Y[i] = g * Cn[i] + (1.f - g) * sdn;
  }
}

// =====================================================================
// Directional GRU scan (correction layer). C=F=16 fixed. In-place on F.
// One persistent workgroup per batch element; weights + 3 row-state
// buffers LDS-resident; __syncthreads between sequential time steps.
// dir: 0 = +W, 1 = -W (lateral axis = H); 2 = +H, 3 = -H (lateral = W).
// =====================================================================
__global__ __launch_bounds__(256)
void k_gru_scan(float* __restrict__ F, const float* __restrict__ wi,
                const float* __restrict__ wh, const float* __restrict__ bi,
                const float* __restrict__ bh, int H, int W, int dir) {
  __shared__ float swi[48 * 48];
  __shared__ float swh[48 * 16];
  __shared__ float sb[96];              // [0:48)=bi, [48:96)=bh
  __shared__ float prev[192 * 16];
  __shared__ float cur[192 * 16];
  __shared__ float nxt[192 * 16];

  const int b = blockIdx.x;
  const int tid = threadIdx.x, nth = blockDim.x;
  const int T = (dir < 2) ? W : H;
  const int L = (dir < 2) ? H : W;
  const bool rev = (dir & 1);

  for (int i = tid; i < 48 * 48; i += nth) swi[i] = wi[i];
  for (int i = tid; i < 48 * 16; i += nth) swh[i] = wh[i];
  if (tid < 48) { sb[tid] = bi[tid]; sb[48 + tid] = bh[tid]; }

  auto gidx = [&](int c, int l, int t) -> size_t {
    int y = (dir < 2) ? l : t;
    int x = (dir < 2) ? t : l;
    return (((size_t)b * 16 + c) * H + y) * W + x;
  };

  int t0 = rev ? T - 1 : 0;
  for (int i = tid; i < L * 16; i += nth)
    prev[i] = F[gidx(i & 15, i >> 4, t0)];
  __syncthreads();

  for (int s = 1; s < T; ++s) {
    int t = rev ? (T - 1 - s) : s;
    for (int i = tid; i < L * 16; i += nth)
      cur[i] = F[gidx(i & 15, i >> 4, t)];
    __syncthreads();

    for (int i = tid; i < L * 16; i += nth) {
      int l = i >> 4, c = i & 15;
      float ir = sb[c], iz = sb[16 + c], in_ = sb[32 + c];
      const float* w0 = &swi[c * 48];
      const float* w1 = &swi[(16 + c) * 48];
      const float* w2 = &swi[(32 + c) * 48];
#pragma unroll 4
      for (int j = 0; j < 16; ++j) {
        float xm = (l > 0)     ? prev[(l - 1) * 16 + j] : 0.f;  // up neighbor
        float xc =               prev[l * 16 + j];              // center
        float xp = (l < L - 1) ? prev[(l + 1) * 16 + j] : 0.f;  // down neighbor
        ir  += xm * w0[j] + xc * w0[16 + j] + xp * w0[32 + j];
        iz  += xm * w1[j] + xc * w1[16 + j] + xp * w1[32 + j];
        in_ += xm * w2[j] + xc * w2[16 + j] + xp * w2[32 + j];
      }
      float hr = sb[48 + c], hz = sb[64 + c], hn = sb[80 + c];
      const float* v0 = &swh[c * 16];
      const float* v1 = &swh[(16 + c) * 16];
      const float* v2 = &swh[(32 + c) * 16];
#pragma unroll 4
      for (int j = 0; j < 16; ++j) {
        float hj = cur[l * 16 + j];
        hr += hj * v0[j]; hz += hj * v1[j]; hn += hj * v2[j];
      }
      float r = 1.f / (1.f + expf(-(ir + hr)));
      float z = 1.f / (1.f + expf(-(iz + hz)));
      float nn = tanhf(in_ + r * hn);
      nxt[i] = (1.f - z) * nn + z * cur[i];
    }
    __syncthreads();

    for (int i = tid; i < L * 16; i += nth) {
      float v = nxt[i];
      prev[i] = v;
      F[gidx(i & 15, i >> 4, t)] = v;
    }
    __syncthreads();
  }
}

// =====================================================================
// Host-side orchestration
// =====================================================================
struct Buf  { float* p; int C, H, W; };
struct ConvP{ const float *w, *b; };
struct BnP  { const float *g, *b; };
struct GruP { const float *wi, *wh, *bi, *bh; };
struct SdnP { ConvP pin; GruP cells[4]; ConvP pout; };
struct ResP { SdnP sdn; ConvP cnn; };
struct VspP { ConvP c1; BnP bn1; ResP sp; BnP bn2; };
struct VggP { ConvP c1; BnP bn1; ConvP c2; BnP bn2; };

extern "C" void kernel_launch(void* const* d_in, const int* in_sizes, int n_in,
                              void* d_out, int out_size, void* d_ws, size_t ws_size,
                              hipStream_t stream) {
  (void)in_sizes; (void)n_in; (void)out_size; (void)ws_size;
  const int B = 2;

  // ---- deterministic bump allocator over d_ws ----
  char* base = (char*)d_ws;
  size_t off = 0;
  auto allocB = [&](size_t bytes) -> void* {
    void* p = base + off;
    off = (off + bytes + 255) & ~(size_t)255;
    return p;
  };
  auto allocF    = [&](size_t n) -> float*  { return (float*)allocB(n * 4); };
  auto allocHalf = [&](size_t n) -> half_t* { return (half_t*)allocB(n * 2); };

  // persistent f16 GEMM staging (worst case: s04 convs, Kp=448, Ncol=73728)
  half_t* Ah = allocHalf((size_t)1 << 20);
  half_t* Bh = allocHalf((size_t)34 << 20);

  auto newbuf = [&](int C, int H, int W) -> Buf {
    return Buf{allocF((size_t)B * C * H * W), C, H, W};
  };
  auto blocksFor = [&](size_t tot) -> int {
    size_t bl = (tot + 255) / 256;
    return (int)(bl > 4096 ? 4096 : bl);
  };

  // ---- parameter cursor: depth-first, dict-insertion order of setup_inputs ----
  int pc = 1;  // d_in[0] = x
  auto P      = [&]() -> const float* { return (const float*)d_in[pc++]; };
  auto rdConv = [&]() { ConvP c; c.w = P(); c.b = P(); return c; };
  auto rdBn   = [&]() { BnP c; c.g = P(); c.b = P(); return c; };
  auto rdGru  = [&]() { GruP g; g.wi = P(); g.wh = P(); g.bi = P(); g.bh = P(); return g; };
  auto rdSdn  = [&]() { SdnP s; s.pin = rdConv(); for (int i = 0; i < 4; ++i) s.cells[i] = rdGru(); s.pout = rdConv(); return s; };
  auto rdRes  = [&]() { ResP r; r.sdn = rdSdn(); r.cnn = rdConv(); return r; };
  auto rdVsp  = [&]() { VspP v; v.c1 = rdConv(); v.bn1 = rdBn(); v.sp = rdRes(); v.bn2 = rdBn(); return v; };
  auto rdVgg  = [&]() { VggP v; v.c1 = rdConv(); v.bn1 = rdBn(); v.c2 = rdConv(); v.bn2 = rdBn(); return v; };

  // ---- primitive ops ----
  auto convOp = [&](Buf x, ConvP p, int Cout, int ks, int pad, int transposed,
                    int act, float* outp) -> Buf {
    int Kp = ((x.C * ks * ks) + 31) & ~31;
    int Mp = (Cout + 15) & ~15;
    size_t Ncol = (size_t)B * x.H * x.W;
    k_wrepack<<<dim3(64), dim3(256), 0, stream>>>(p.w, Ah, Cout, Mp, x.C, ks, Kp, transposed);
    k_im2col<<<dim3(blocksFor(Ncol * Kp)), dim3(256), 0, stream>>>(
        x.p, Bh, B, x.C, x.H, x.W, ks, pad, Kp);
    Buf y{outp ? outp : allocF((size_t)B * Cout * x.H * x.W), Cout, x.H, x.W};
    int units = (Mp >> 4) * ((int)Ncol >> 5);     // waves: 16x32 slab each
    k_gemm_wmma<<<dim3((units + 7) / 8), dim3(256), 0, stream>>>(
        Ah, Bh, y.p, p.b, Cout, (int)Ncol, x.H * x.W, Kp, act);
    return y;
  };

  auto bnOp = [&](Buf x, BnP p, int relu) -> Buf {
    float* mv = allocF(2 * (size_t)x.C);
    k_bn_stats<<<dim3(x.C), dim3(256), 0, stream>>>(x.p, mv, mv + x.C, B, x.C, x.H * x.W);
    Buf y = newbuf(x.C, x.H, x.W);
    size_t tot = (size_t)B * x.C * x.H * x.W;
    k_bn_apply<<<dim3(blocksFor(tot)), dim3(256), 0, stream>>>(
        x.p, y.p, mv, mv + x.C, p.g, p.b, B, x.C, x.H * x.W, relu);
    return y;
  };

  auto sdnOp = [&](Buf x, SdnP& s, int oc2) -> Buf {
    // project_in is ALWAYS ConvTranspose2d (source bug), tanh fused in epilogue
    Buf h = convOp(x, s.pin, 16, 3, 1, /*transposed=*/1, /*act=tanh*/2, nullptr);
    for (int d = 0; d < 4; ++d)
      k_gru_scan<<<dim3(B), dim3(256), 0, stream>>>(
          h.p, s.cells[d].wi, s.cells[d].wh, s.cells[d].bi, s.cells[d].bh, h.H, h.W, d);
    return convOp(h, s.pout, oc2, 1, 0, 0, 0, nullptr);
  };

  auto resOp = [&](Buf x, ResP& r, int oc, int up) -> Buf {
    Buf cnn = convOp(x, r.cnn, oc, 3, 1, up ? 1 : 0, 0, nullptr);
    Buf sf = sdnOp(x, r.sdn, 2 * oc);
    Buf y = newbuf(oc, x.H, x.W);
    size_t tot = (size_t)B * oc * x.H * x.W;
    k_gate<<<dim3(blocksFor(tot)), dim3(256), 0, stream>>>(sf.p, cnn.p, y.p, B, oc, x.H * x.W);
    return y;
  };

  auto vspOp = [&](Buf x, VspP& v, int mc, int oc, int up) -> Buf {
    Buf h = bnOp(convOp(x, v.c1, mc, 3, 1, 0, 0, nullptr), v.bn1, 1);
    h = resOp(h, v.sp, oc, up);
    return bnOp(h, v.bn2, 1);
  };

  auto vggOp = [&](Buf x, VggP& v, int mc, int oc) -> Buf {
    Buf h = bnOp(convOp(x, v.c1, mc, 3, 1, 0, 0, nullptr), v.bn1, 1);
    return bnOp(convOp(h, v.c2, oc, 3, 1, 0, 0, nullptr), v.bn2, 1);
  };

  auto poolOp = [&](Buf x) -> Buf {
    Buf y = newbuf(x.C, x.H / 2, x.W / 2);
    size_t tot = (size_t)B * x.C * y.H * y.W;
    k_maxpool2<<<dim3(blocksFor(tot)), dim3(256), 0, stream>>>(x.p, y.p, B, x.C, x.H, x.W);
    return y;
  };

  auto upOp = [&](Buf x) -> Buf {
    Buf y = newbuf(x.C, 2 * x.H, 2 * x.W);
    size_t tot = (size_t)B * x.C * y.H * y.W;
    k_up2<<<dim3(blocksFor(tot)), dim3(256), 0, stream>>>(x.p, y.p, B, x.C, x.H, x.W);
    return y;
  };

  auto catOp = [&](Buf a, Buf c) -> Buf {
    Buf y = newbuf(a.C + c.C, a.H, a.W);
    size_t tot = (size_t)B * y.C * a.H * a.W;
    k_concat<<<dim3(blocksFor(tot)), dim3(256), 0, stream>>>(
        a.p, c.p, y.p, B, a.C, c.C, a.H * a.W);
    return y;
  };

  // ---- read params in construction order ----
  VspP s00 = rdVsp(); VggP c10 = rdVgg(); VspP s20 = rdVsp();
  VspP s30 = rdVsp(); VspP s40 = rdVsp(); VspP s31 = rdVsp();
  VspP s22 = rdVsp(); VggP c13 = rdVgg(); VspP s04 = rdVsp();
  ConvP fin = rdConv();

  // ---- UNet++ forward ----
  Buf xin{(float*)d_in[0], 3, 192, 192};
  Buf x00 = vspOp(xin, s00, 16, 16, 0);
  Buf x10 = vggOp(poolOp(x00), c10, 32, 32);
  Buf x20 = vspOp(poolOp(x10), s20, 64, 64, 0);
  Buf x30 = vspOp(poolOp(x20), s30, 128, 128, 0);
  Buf x40 = vspOp(poolOp(x30), s40, 256, 256, 0);
  Buf x31 = vspOp(catOp(x30, upOp(x40)), s31, 128, 128, 1);
  Buf x22 = vspOp(catOp(x20, upOp(x31)), s22, 64, 64, 1);
  Buf x13 = vggOp(catOp(x10, upOp(x22)), c13, 32, 32);
  Buf x04 = vspOp(catOp(x00, upOp(x13)), s04, 16, 16, 1);
  convOp(x04, fin, 1, 1, 0, 0, 0, (float*)d_out);   // final 1x1 conv -> d_out
}
